// MemoryNet_33758442946977
// MI455X (gfx1250) — compile-verified, matrix operand order checked by reading, not run on hardware
//
#include <hip/hip_runtime.h>
#include <hip/hip_bf16.h>

typedef __attribute__((ext_vector_type(16))) __bf16 v16bf;
typedef __attribute__((ext_vector_type(8)))  float  v8f;

#define BM 128      // m rows per workgroup (8 waves x 16)
#define BN 64       // n tile per iteration
#define MT 2048     // m extent (also q/k row stride)
#define NTOT 2048   // n extent
// LDS layout (ushort offsets)
#define KBUF 0      // [nt:4][kc:4][lane:32][hi:16]  B-fragments of K-tile
#define VBUF 8192   // [ct:8][nk:2][lane:32][hi:16]  B-fragments of V^T-tile
#define PBUF 16384  // [wave:8][nk:2][lane:32][hi:16] A-fragments of P

__device__ __forceinline__ unsigned short f2bf(float f) {
  union { __bf16 h; unsigned short u; } x; x.h = (__bf16)f; return x.u;
}

__global__ __launch_bounds__(256, 1) void
fattn_bf16_wmma(const float* __restrict__ q, const float* __restrict__ k,
                const float* __restrict__ v, float* __restrict__ out)
{
  __shared__ __align__(16) unsigned short smem[24576]; // 48 KB

  const int t  = threadIdx.x;
  const int w  = t >> 5;      // wave 0..7
  const int l  = t & 31;      // lane
  const int lm = l & 15;
  const int lg = l >> 4;
  const int m0 = blockIdx.x * BM;
  const int bh = blockIdx.y;  // 0..31
  const size_t plane = ((size_t)(bh >> 3) * 1024 + (size_t)(bh & 7) * 128) * (size_t)MT;
  const float* qg = q + plane;
  const float* kg = k + plane;
  const float* vg = v + plane;
  float*       og = out + plane;

  const float scale = 0.08838834764831845f; // 1/sqrt(128)

  // ---- Q resident in registers as bf16 A-fragments (16 m-rows x dh=128), pre-scaled
  v16bf qfrag[4];
  #pragma unroll
  for (int c = 0; c < 4; ++c) {
    #pragma unroll
    for (int hi = 0; hi < 16; ++hi) {
      int kk = (hi < 8) ? (hi + 8 * lg) : (8 + hi + 8 * lg);
      int dd = c * 32 + kk;
      float val = qg[(size_t)dd * MT + (m0 + 16 * w + lm)] * scale;
      qfrag[c][hi] = (__bf16)val;
    }
  }

  v8f acc[8];
  #pragma unroll
  for (int i = 0; i < 8; ++i) acc[i] = (v8f){};
  float rowmax[8], rowsum[8];
  #pragma unroll
  for (int i = 0; i < 8; ++i) { rowmax[i] = -__builtin_inff(); rowsum[i] = 0.0f; }

  for (int n0 = 0; n0 < NTOT; n0 += BN) {
    // ---- K tile fill: row-pairs (dd, dd+1) pack into one b32 LDS store
    //      (adjacent dd -> same fragment lane, adjacent hi, for even dd)
    #pragma unroll
    for (int i = 0; i < 4; ++i) {
      int u  = t + i * 256;          // 1024 units = 64 row-pairs x 16 float4-cols
      int dd = (u >> 4) << 1;        // even row
      int j  = (u & 15) << 2;        // col within n-tile
      const float4 k0 = *(const float4*)(kg + (size_t)dd * NTOT + n0 + j);
      const float4 k1 = *(const float4*)(kg + (size_t)(dd + 1) * NTOT + n0 + j);
      float k0e[4] = {k0.x, k0.y, k0.z, k0.w};
      float k1e[4] = {k1.x, k1.y, k1.z, k1.w};
      int kc = dd >> 5, kloc = dd & 31;
      int lanehi = 16 * ((kloc >> 3) & 1);
      int hi = (kloc & 7) | ((kloc >> 4) << 3);   // even
      #pragma unroll
      for (int e = 0; e < 4; ++e) {
        int nl = j + e;
        int nt = nl >> 4;
        int lane = (nl & 15) + lanehi;
        unsigned pv = (unsigned)f2bf(k0e[e]) | ((unsigned)f2bf(k1e[e]) << 16);
        *(unsigned*)&smem[KBUF + (((nt * 4 + kc) * 32 + lane) << 4) + hi] = pv;
      }
    }
    // ---- V tile fill: column-pairs (nl, nl+1) pack into one b32 LDS store
    #pragma unroll
    for (int i = 0; i < 8; ++i) {
      int u  = t + i * 256;          // 2048 units = 128 rows x 16 float4-cols
      int cc = u >> 4;
      int j  = (u & 15) << 2;
      const float4 vq = *(const float4*)(vg + (size_t)cc * NTOT + n0 + j);
      float ve[4] = {vq.x, vq.y, vq.z, vq.w};
      int ct = cc >> 4;
      int Nl = cc & 15;
      #pragma unroll
      for (int e2 = 0; e2 < 2; ++e2) {
        int nl = j + e2 * 2;         // even
        int nk = nl >> 5, kloc = nl & 31;
        int lane = Nl + 16 * ((kloc >> 3) & 1);
        int hi = (kloc & 7) | ((kloc >> 4) << 3);   // even
        unsigned pv = (unsigned)f2bf(ve[e2 * 2]) | ((unsigned)f2bf(ve[e2 * 2 + 1]) << 16);
        *(unsigned*)&smem[VBUF + (((ct * 2 + nk) * 32 + lane) << 4) + hi] = pv;
      }
    }
    __syncthreads();

    // ---- S = scaled Q^T K : kc outer so A (qfrag) stays stationary across 4 WMMAs
    v8f s[4];
    #pragma unroll
    for (int nt = 0; nt < 4; ++nt) s[nt] = (v8f){};
    #pragma unroll
    for (int kc = 0; kc < 4; ++kc) {
      #pragma unroll
      for (int nt = 0; nt < 4; ++nt) {
        v16bf kf = *(const v16bf*)&smem[KBUF + (((nt * 4 + kc) * 32 + l) << 4)];
        s[nt] = __builtin_amdgcn_wmma_f32_16x16x32_bf16(false, qfrag[kc], false, kf,
                                                        (short)0, s[nt], false, false);
      }
    }

    // ---- online softmax: row max (cross-lane within 16-lane half), rescale, exp
    float mnew[8], corr[8];
    #pragma unroll
    for (int vv = 0; vv < 8; ++vv) {
      float tm = s[0][vv];
      #pragma unroll
      for (int nt = 1; nt < 4; ++nt) tm = fmaxf(tm, s[nt][vv]);
      #pragma unroll
      for (int xm = 1; xm <= 8; xm <<= 1) tm = fmaxf(tm, __shfl_xor(tm, xm));
      mnew[vv] = fmaxf(rowmax[vv], tm);
      corr[vv] = __expf(rowmax[vv] - mnew[vv]);
    }

    float psum[8];
    #pragma unroll
    for (int vv = 0; vv < 8; ++vv) psum[vv] = 0.0f;

    // P = exp(S - mnew), written straight into A-fragment order in LDS
    #pragma unroll
    for (int nt = 0; nt < 4; ++nt) {
      #pragma unroll
      for (int vv = 0; vv < 8; ++vv) {
        float p = __expf(s[nt][vv] - mnew[vv]);
        psum[vv] += p;
        int M  = vv + 8 * lg;            // C-layout row this lane holds
        int tl = M + 16 * (lm >> 3);     // target lane in A layout
        int hi = (lm & 7) | ((nt & 1) << 3);
        smem[PBUF + (((w * 2 + (nt >> 1)) * 32 + tl) << 4) + hi] = f2bf(p);
      }
    }

    #pragma unroll
    for (int vv = 0; vv < 8; ++vv) {
      float ps = psum[vv];
      #pragma unroll
      for (int xm = 1; xm <= 8; xm <<= 1) ps += __shfl_xor(ps, xm);
      rowsum[vv] = rowsum[vv] * corr[vv] + ps;
      rowmax[vv] = mnew[vv];
    }
    #pragma unroll
    for (int ct = 0; ct < 8; ++ct)
      #pragma unroll
      for (int vv = 0; vv < 8; ++vv) acc[ct][vv] *= corr[vv];

    // ---- O += P * V^T : nk outer so A (pf) stays stationary across 8 WMMAs
    asm volatile("s_wait_dscnt 0" ::: "memory");
    #pragma unroll
    for (int nk = 0; nk < 2; ++nk) {
      v16bf pf = *(const v16bf*)&smem[PBUF + (((w * 2 + nk) * 32 + l) << 4)];
      #pragma unroll
      for (int ct = 0; ct < 8; ++ct) {
        v16bf vf = *(const v16bf*)&smem[VBUF + (((ct * 2 + nk) * 32 + l) << 4)];
        acc[ct] = __builtin_amdgcn_wmma_f32_16x16x32_bf16(false, pf, false, vf,
                                                          (short)0, acc[ct], false, false);
      }
    }
    __syncthreads();
  }

  // ---- normalize by softmax denominator
  #pragma unroll
  for (int vv = 0; vv < 8; ++vv) {
    float rinv = __frcp_rn(rowsum[vv]);
    #pragma unroll
    for (int ct = 0; ct < 8; ++ct) acc[ct][vv] *= rinv;
  }

  // ---- epilogue: LDS transpose (2 passes of 64 channels), coalesced float4 stores
  //      each lane's 8 acc rows are consecutive m -> two b128 LDS stores per tile
  float* ob = (float*)smem; // 64 x 128 f32 = 32 KB, reuses tile buffers
  #pragma unroll
  for (int hc = 0; hc < 2; ++hc) {
    __syncthreads();
    #pragma unroll
    for (int ct = 0; ct < 4; ++ct) {
      v8f a = acc[hc * 4 + ct];
      float4 lo4 = make_float4(a[0], a[1], a[2], a[3]);
      float4 hi4 = make_float4(a[4], a[5], a[6], a[7]);
      float* dst = &ob[(ct * 16 + lm) * BM + 16 * w + 8 * lg];
      *(float4*)(dst)     = lo4;
      *(float4*)(dst + 4) = hi4;
    }
    __syncthreads();
    #pragma unroll
    for (int i = 0; i < 8; ++i) {
      int u  = t + i * 256;     // 2048 float4 units = 64 rows x 32 float4
      int cl = u >> 5;          // channel 0..63
      int mm = (u & 31) << 2;   // m offset 0..124
      float4 val = *(const float4*)&ob[cl * BM + mm];
      *(float4*)(og + (size_t)(hc * 64 + cl) * MT + m0 + mm) = val;
    }
  }
}

extern "C" void kernel_launch(void* const* d_in, const int* in_sizes, int n_in,
                              void* d_out, int out_size, void* d_ws, size_t ws_size,
                              hipStream_t stream) {
  (void)in_sizes; (void)n_in; (void)d_ws; (void)ws_size; (void)out_size;
  const float* q = (const float*)d_in[0];
  const float* k = (const float*)d_in[1];
  const float* v = (const float*)d_in[2];
  float* out = (float*)d_out;
  dim3 grid(MT / BM, 32); // 16 m-tiles x 32 batch-heads
  fattn_bf16_wmma<<<grid, 256, 0, stream>>>(q, k, v, out);
}
